// MultiHeadAttention_90967407329635
// MI455X (gfx1250) — matrix operand order repaired and checked by reading, not compiled, not run on hardware
//
#include <hip/hip_runtime.h>

typedef __attribute__((ext_vector_type(16))) __bf16 v16bf;
typedef __attribute__((ext_vector_type(8)))  __bf16 v8bf;
typedef __attribute__((ext_vector_type(8)))  float  v8f;
typedef __attribute__((ext_vector_type(4)))  int    v4i;

#define NEGF (-4294967295.0f)
#define NB   32
#define NL   1024
#define ND   512
#define NH   8
#define DEPTH 64

// ---- CDNA5 async global->LDS path (ASYNCcnt), guarded with fallback --------
#if defined(__has_builtin)
#if __has_builtin(__builtin_amdgcn_global_load_async_to_lds_b128)
#define USE_ASYNC_LDS 1
#endif
#endif
#ifndef USE_ASYNC_LDS
#define USE_ASYNC_LDS 0
#endif

__device__ __forceinline__ void async_cp16(const __bf16* g, __bf16* l) {
#if USE_ASYNC_LDS
  // param1: addrspace(1) v4i*  (printed by clang as "vector_size(16) int __device__ *")
  // param2: addrspace(3) v4i*  ; param3/4: imm offset, imm cpol
  __attribute__((address_space(1))) v4i* gp =
      (__attribute__((address_space(1))) v4i*)((void*)g);
  __attribute__((address_space(3))) v4i* lp =
      (__attribute__((address_space(3))) v4i*)((void*)l);
  __builtin_amdgcn_global_load_async_to_lds_b128(gp, lp, 0, 0);
#else
  *(v8bf*)l = *(const v8bf*)g;
#endif
}

__device__ __forceinline__ void wait_async_lds() {
#if USE_ASYNC_LDS
#if __has_builtin(__builtin_amdgcn_s_wait_asynccnt)
  __builtin_amdgcn_s_wait_asynccnt(0);
#else
  asm volatile("s_wait_asynccnt 0x0" ::: "memory");
#endif
#endif
}

__device__ __forceinline__ v16bf cat8(v8bf lo, v8bf hi) {
  return __builtin_shufflevector(lo, hi, 0,1,2,3,4,5,6,7,8,9,10,11,12,13,14,15);
}

__device__ __forceinline__ v8f wmma_bf16(v16bf a, v16bf b, v8f c) {
  // v_wmma_f32_16x16x32_bf16: D = A(16x32) * B(32x16) + C, f32 accum
  return __builtin_amdgcn_wmma_f32_16x16x32_bf16(false, a, false, b, (short)0, c, false, false);
}

// ---------------------------------------------------------------------------
// Kernel 1: fused QKV projection.  out = X @ W^T + b   (M=32768, N=512, K=512)
// grid.z: 0 -> Q from queries/Wq, 1 -> K from keys/Wk, 2 -> V from keys/Wv
// Q,K stored bf16 as [b, h, l, d]; V stored bf16 transposed as [b, h, d, l].
// Double-buffered LDS staging: prefetch k0+32 while running WMMAs on k0.
// ---------------------------------------------------------------------------
__global__ __launch_bounds__(256) void qkv_gemm(
    const float* __restrict__ qin, const float* __restrict__ kin,
    const float* __restrict__ Wq, const float* __restrict__ bq,
    const float* __restrict__ Wk, const float* __restrict__ bk,
    const float* __restrict__ Wv, const float* __restrict__ bv,
    unsigned short* __restrict__ Qh_, unsigned short* __restrict__ Kh_,
    unsigned short* __restrict__ Vt_)
{
  __shared__ __bf16 As[2][128][40];   // 128 x 32 tile of X (bf16), +8 pad
  __shared__ __bf16 Bs[2][128][40];   // 128 x 32 tile of W rows (bf16)

  const int mode = blockIdx.z;
  const float* X    = (mode == 0) ? qin : kin;
  const float* W    = (mode == 0) ? Wq : (mode == 1 ? Wk : Wv);
  const float* bias = (mode == 0) ? bq : (mode == 1 ? bk : bv);

  const int m0 = blockIdx.y * 128;
  const int n0 = blockIdx.x * 128;
  const int tid  = threadIdx.x;
  const int wid  = tid >> 5, lane = tid & 31;
  const int wm   = wid >> 1, wn   = wid & 1;       // 4x2 wave grid, 32x64 each
  const int lh   = lane & 15;
  const int hiL  = (lane >> 4) & 1;

  const int lr = tid >> 3;        // row within 32-row pass
  const int lc = (tid & 7) * 4;   // 4-float column chunk

  auto stage = [&](int k0, int bufi) {
    #pragma unroll
    for (int p = 0; p < 4; p++) {
      int row = lr + p * 32;
      float4 xa = *(const float4*)(X + (size_t)(m0 + row) * ND + k0 + lc);
      As[bufi][row][lc + 0] = (__bf16)xa.x; As[bufi][row][lc + 1] = (__bf16)xa.y;
      As[bufi][row][lc + 2] = (__bf16)xa.z; As[bufi][row][lc + 3] = (__bf16)xa.w;
      float4 wb = *(const float4*)(W + (size_t)(n0 + row) * ND + k0 + lc);
      Bs[bufi][row][lc + 0] = (__bf16)wb.x; Bs[bufi][row][lc + 1] = (__bf16)wb.y;
      Bs[bufi][row][lc + 2] = (__bf16)wb.z; Bs[bufi][row][lc + 3] = (__bf16)wb.w;
    }
  };

  v8f acc[2][4];
  #pragma unroll
  for (int i = 0; i < 2; i++)
    #pragma unroll
    for (int j = 0; j < 4; j++)
      #pragma unroll
      for (int e = 0; e < 8; e++) acc[i][j][e] = 0.0f;

  stage(0, 0);
  __syncthreads();

  int buf = 0;
  for (int k0 = 0; k0 < ND; k0 += 32) {
    if (k0 + 32 < ND) stage(k0 + 32, buf ^ 1);   // overlap with WMMAs below

    v16bf af[2], bfm[4];
    #pragma unroll
    for (int i = 0; i < 2; i++) {        // A fragments (16-bit A layout, 16x32)
      int ar = wm * 32 + i * 16 + lh;
      int ab = hiL ? 8 : 0;
      v8bf lo = *(const v8bf*)&As[buf][ar][ab];
      v8bf hi = *(const v8bf*)&As[buf][ar][ab + 16];
      af[i] = cat8(lo, hi);
    }
    #pragma unroll
    for (int j = 0; j < 4; j++) {        // B fragments (32x16): same k order as A
      int br = wn * 64 + j * 16 + lh;
      int kb = hiL ? 8 : 0;
      v8bf lo = *(const v8bf*)&Bs[buf][br][kb];
      v8bf hi = *(const v8bf*)&Bs[buf][br][kb + 16];
      bfm[j] = cat8(lo, hi);
    }
    #pragma unroll
    for (int i = 0; i < 2; i++)
      #pragma unroll
      for (int j = 0; j < 4; j++)
        acc[i][j] = wmma_bf16(af[i], bfm[j], acc[i][j]);

    __syncthreads();
    buf ^= 1;
  }

  __bf16* Qh = (__bf16*)Qh_;
  __bf16* Kh = (__bf16*)Kh_;
  __bf16* Vt = (__bf16*)Vt_;

  #pragma unroll
  for (int j = 0; j < 4; j++) {
    int n = n0 + wn * 64 + j * 16 + lh;     // output feature
    int h = n >> 6, d = n & 63;
    float bb = bias[n];
    #pragma unroll
    for (int i = 0; i < 2; i++) {
      int mrow = m0 + wm * 32 + i * 16 + (hiL ? 8 : 0);
      if (mode == 2) {
        // V transposed: [b,h,d,l]; 8 consecutive rows -> contiguous along l
        int b_ = mrow >> 10, l_ = mrow & 1023;
        v8bf pk;
        #pragma unroll
        for (int e = 0; e < 8; e++) pk[e] = (__bf16)(acc[i][j][e] + bb);
        *(v8bf*)(Vt + (((size_t)b_ * NH + h) * DEPTH + d) * NL + l_) = pk;
      } else {
        __bf16* dst = (mode == 0) ? Qh : Kh;
        #pragma unroll
        for (int e = 0; e < 8; e++) {
          int m = mrow + e;
          int b_ = m >> 10, l_ = m & 1023;
          dst[(((size_t)b_ * NH + h) * NL + l_) * DEPTH + d] = (__bf16)(acc[i][j][e] + bb);
        }
      }
    }
  }
}

// ---------------------------------------------------------------------------
// Kernel 2: padding masks.  mask[b,l] = (sum over 512 features != 0) ? 1 : 0
// ---------------------------------------------------------------------------
__global__ __launch_bounds__(256) void row_mask(
    const float* __restrict__ qin, const float* __restrict__ kin,
    float* __restrict__ qmask, float* __restrict__ kmask)
{
  int row  = blockIdx.x * 8 + (threadIdx.x >> 5);
  int lane = threadIdx.x & 31;
  float s1 = 0.0f, s2 = 0.0f;
  const size_t base = (size_t)row * ND;
  for (int i = lane; i < ND; i += 32) {
    s1 += qin[base + i];
    s2 += kin[base + i];
  }
  #pragma unroll
  for (int off = 16; off > 0; off >>= 1) {
    s1 += __shfl_xor(s1, off, 32);
    s2 += __shfl_xor(s2, off, 32);
  }
  if (lane == 0) {
    qmask[row] = (s1 != 0.0f) ? 1.0f : 0.0f;
    kmask[row] = (s2 != 0.0f) ? 1.0f : 0.0f;
  }
}

// ---------------------------------------------------------------------------
// Kernel 3: flash attention per (q-tile of 64, head, batch).
// 4 waves x 16 q-rows.  Causal: only key tiles kt <= qt.
// K/V tiles double-buffered in LDS and prefetched with async global->LDS.
// ---------------------------------------------------------------------------
__global__ __launch_bounds__(128) void attn(
    const unsigned short* __restrict__ Qh_, const unsigned short* __restrict__ Kh_,
    const unsigned short* __restrict__ Vt_,
    const float* __restrict__ qmask, const float* __restrict__ kmask,
    const float* __restrict__ qin, float* __restrict__ out)
{
  __shared__ __bf16 Ks[2][64][72];    // K tile: [key][d]
  __shared__ __bf16 Vs[2][64][72];    // V tile transposed: [d][key]
  __shared__ __bf16 Pm[4][16][64];    // per-wave P scratch for C->A relayout

  const __bf16* Qh = (const __bf16*)Qh_;
  const __bf16* Kh = (const __bf16*)Kh_;
  const __bf16* Vt = (const __bf16*)Vt_;

  const int qt = blockIdx.x, h = blockIdx.y, b = blockIdx.z;
  const int tid = threadIdx.x;
  const int wid = tid >> 5, lane = tid & 31;
  const int lh = lane & 15;
  const int hiL = (lane >> 4) & 1;
  const int abase = hiL ? 8 : 0;
  const size_t bh = (size_t)b * NH + h;
  const int q0 = qt * 64 + wid * 16;
  const int rbase = q0 + (hiL ? 8 : 0);

  auto load_tiles = [&](int kt, int bufi) {   // 128 threads, 4x16B each per tile
    int r  = tid >> 1;
    int c0 = (tid & 1) * 32;
    const __bf16* kp = Kh + (bh * NL + kt * 64 + r) * DEPTH + c0;
    const __bf16* vp = Vt + (bh * DEPTH + r) * NL + kt * 64 + c0;
    #pragma unroll
    for (int c = 0; c < 4; c++) {
      async_cp16(kp + c * 8, &Ks[bufi][r][c0 + c * 8]);
      async_cp16(vp + c * 8, &Vs[bufi][r][c0 + c * 8]);
    }
  };

  // preload the wave's Q strip (16 rows x 64) as two A-fragments (k-steps)
  v16bf qf[2];
  {
    const __bf16* qp = Qh + (bh * NL + q0 + lh) * DEPTH;
    #pragma unroll
    for (int s = 0; s < 2; s++) {
      v8bf lo = *(const v8bf*)(qp + s * 32 + abase);
      v8bf hi = *(const v8bf*)(qp + s * 32 + abase + 16);
      qf[s] = cat8(lo, hi);
    }
  }

  v8f o[4];
  #pragma unroll
  for (int j = 0; j < 4; j++)
    #pragma unroll
    for (int e = 0; e < 8; e++) o[j][e] = 0.0f;
  float mrun[8], lrun[8];
  #pragma unroll
  for (int e = 0; e < 8; e++) { mrun[e] = -3.0e38f; lrun[e] = 0.0f; }

  load_tiles(0, 0);
  wait_async_lds();
  __syncthreads();

  int buf = 0;
  for (int kt = 0; kt <= qt; ++kt) {
    if (kt < qt) load_tiles(kt + 1, buf ^ 1);   // async prefetch next tile

    // S = Q K^T  (B-matrix fragment for key col n = row n of Ks, same k order as qf)
    v8f sacc[4];
    #pragma unroll
    for (int j = 0; j < 4; j++)
      #pragma unroll
      for (int e = 0; e < 8; e++) sacc[j][e] = 0.0f;
    #pragma unroll
    for (int s = 0; s < 2; s++) {
      int kb = s * 32 + (hiL ? 8 : 0);
      #pragma unroll
      for (int j = 0; j < 4; j++) {
        int n = j * 16 + lh;
        v8bf lo = *(const v8bf*)&Ks[buf][n][kb];
        v8bf hi = *(const v8bf*)&Ks[buf][n][kb + 16];
        sacc[j] = wmma_bf16(qf[s], cat8(lo, hi), sacc[j]);
      }
    }

    // scale + causal + key-padding mask
    float kmv[4];
    #pragma unroll
    for (int j = 0; j < 4; j++) kmv[j] = kmask[b * NL + kt * 64 + j * 16 + lh];
    float p[4][8], tmax[8];
    #pragma unroll
    for (int e = 0; e < 8; e++) {
      int qrow = rbase + e;
      float mx = -3.0e38f;
      #pragma unroll
      for (int j = 0; j < 4; j++) {
        int key = kt * 64 + j * 16 + lh;
        float v = sacc[j][e] * 0.125f;           // 1/sqrt(64)
        if (key > qrow || kmv[j] == 0.0f) v = NEGF;
        p[j][e] = v;
        mx = fmaxf(mx, v);
      }
      #pragma unroll
      for (int off = 1; off < 16; off <<= 1) mx = fmaxf(mx, __shfl_xor(mx, off, 32));
      tmax[e] = mx;
    }

    // online softmax update
    float alpha[8];
    #pragma unroll
    for (int e = 0; e < 8; e++) {
      float mnew = fmaxf(mrun[e], tmax[e]);
      alpha[e] = __expf(mrun[e] - mnew);
      mrun[e] = mnew;
      float rs = 0.0f;
      #pragma unroll
      for (int j = 0; j < 4; j++) {
        float pe = __expf(p[j][e] - mnew);
        p[j][e] = pe;
        rs += pe;
      }
      #pragma unroll
      for (int off = 1; off < 16; off <<= 1) rs += __shfl_xor(rs, off, 32);
      lrun[e] = lrun[e] * alpha[e] + rs;
    }

    // P: C-layout -> A-layout via per-wave LDS scratch (same-wave dep only)
    #pragma unroll
    for (int j = 0; j < 4; j++)
      #pragma unroll
      for (int e = 0; e < 8; e++)
        Pm[wid][(hiL ? 8 : 0) + e][j * 16 + lh] = (__bf16)p[j][e];
    v16bf pf[2];
    {
      const __bf16* pb = &Pm[wid][lh][0];
      #pragma unroll
      for (int s = 0; s < 2; s++) {
        v8bf lo = *(const v8bf*)(pb + s * 32 + abase);
        v8bf hi = *(const v8bf*)(pb + s * 32 + abase + 16);
        pf[s] = cat8(lo, hi);
      }
    }

    // O = O*alpha + P V   (B fragment for depth col n = row n of Vs)
    #pragma unroll
    for (int j = 0; j < 4; j++)
      #pragma unroll
      for (int e = 0; e < 8; e++) o[j][e] *= alpha[e];
    #pragma unroll
    for (int s = 0; s < 2; s++) {
      int kb = s * 32 + (hiL ? 8 : 0);
      #pragma unroll
      for (int j = 0; j < 4; j++) {
        int n = j * 16 + lh;
        v8bf lo = *(const v8bf*)&Vs[buf][n][kb];
        v8bf hi = *(const v8bf*)&Vs[buf][n][kb + 16];
        o[j] = wmma_bf16(pf[s], cat8(lo, hi), o[j]);
      }
    }

    wait_async_lds();      // prefetch complete before the buffer swap barrier
    __syncthreads();
    buf ^= 1;
  }

  // epilogue: normalize, query mask, residual
  float inv[8], qmv[8];
  #pragma unroll
  for (int e = 0; e < 8; e++) {
    int qr = rbase + e;
    inv[e] = (lrun[e] > 0.0f) ? (1.0f / lrun[e]) : 0.0f;
    qmv[e] = qmask[b * NL + qr];
  }
  #pragma unroll
  for (int j = 0; j < 4; j++) {
    int d = j * 16 + lh;
    #pragma unroll
    for (int e = 0; e < 8; e++) {
      int qr = rbase + e;
      size_t oidx = ((size_t)b * NL + qr) * ND + h * DEPTH + d;
      out[oidx] = o[j][e] * inv[e] * qmv[e] + qin[oidx];
    }
  }
}

// ---------------------------------------------------------------------------
extern "C" void kernel_launch(void* const* d_in, const int* in_sizes, int n_in,
                              void* d_out, int out_size, void* d_ws, size_t ws_size,
                              hipStream_t stream) {
  const float* qin = (const float*)d_in[0];
  const float* kin = (const float*)d_in[1];
  const float* Wq  = (const float*)d_in[2];
  const float* bq  = (const float*)d_in[3];
  const float* Wk  = (const float*)d_in[4];
  const float* bk  = (const float*)d_in[5];
  const float* Wv  = (const float*)d_in[6];
  const float* bv  = (const float*)d_in[7];
  float* out = (float*)d_out;

  char* ws = (char*)d_ws;
  const size_t QB = (size_t)NB * NH * NL * DEPTH * sizeof(unsigned short); // 32 MB
  unsigned short* Qh = (unsigned short*)(ws);
  unsigned short* Kh = (unsigned short*)(ws + QB);
  unsigned short* Vt = (unsigned short*)(ws + 2 * QB);
  float* qm = (float*)(ws + 3 * QB);
  float* km = (float*)(ws + 3 * QB + (size_t)NB * NL * sizeof(float));

  // 1) QKV projection: grid (Ntiles=4, Mtiles=256, 3 outputs)
  qkv_gemm<<<dim3(4, 256, 3), 256, 0, stream>>>(qin, kin, Wq, bq, Wk, bk, Wv, bv,
                                                Qh, Kh, Vt);
  // 2) padding masks
  row_mask<<<dim3(NB * NL / 8), 256, 0, stream>>>(qin, kin, qm, km);
  // 3) attention: grid (q-tiles=16, heads=8, batch=32)
  attn<<<dim3(NL / 64, NH, NB), 128, 0, stream>>>(Qh, Kh, Vt, qm, km, qin, out);
}
